// Experts_66795331387840
// MI455X (gfx1250) — matrix operand order repaired and checked by reading, not compiled, not run on hardware
//
#include <hip/hip_runtime.h>
#include <hip/hip_bf16.h>
#include <stdint.h>

// Problem constants (match reference setup_inputs()).
#define HIDDEN 2048
#define FFN    1408
#define FFN2   2816   // 2*FFN
#define NE     8

// Tiling.
#define TM  128   // rows per block
#define TN  64    // cols per block
#define KT  32    // K per iteration (bf16 WMMA K)
#define LDA 40    // padded LDS row stride (halves): 80B, 16B aligned, conflict-friendly
#define LDB 40

typedef __attribute__((ext_vector_type(16))) __bf16 v16bf;
typedef __attribute__((ext_vector_type(8)))  __bf16 v8bf;
typedef __attribute__((ext_vector_type(8)))  float  v8f;
typedef __attribute__((ext_vector_type(4)))  unsigned int u32x4;
typedef __attribute__((ext_vector_type(8)))  int i32x8;
typedef __attribute__((ext_vector_type(4)))  int i32x4;

__device__ __forceinline__ unsigned short f2bfu(float f) {
  union { float f; unsigned u; } cv; cv.f = f;
  unsigned u = cv.u;
  // round-to-nearest-even fp32 -> bf16
  return (unsigned short)((u + 0x7FFFu + ((u >> 16) & 1u)) >> 16);
}

// LDS byte offset of a shared-memory object (AS3 addresses start at 0).
__device__ __forceinline__ unsigned lds_offset_of(const void* p) {
  return (unsigned)(uintptr_t)(__attribute__((address_space(3))) const void*)p;
}

// A fragment (16x32 bf16, M x K) per CDNA5 ISA layout:
// lane L: m = L&15, half = L>>4; elems 0..7 -> K = 8*half + i ; elems 8..15 -> K = 8*half + 16 + i
__device__ __forceinline__ v16bf frag_a(const __bf16* s /* &As[mb*LDA] */, int lane) {
  const int m = lane & 15, half = lane >> 4;
  const __bf16* p = s + m * LDA + half * 8;
  v8bf lo = *(const v8bf*)(p);
  v8bf hi = *(const v8bf*)(p + 16);
  v16bf a;
#pragma unroll
  for (int i = 0; i < 8; ++i) { a[i] = lo[i]; a[i + 8] = hi[i]; }
  return a;
}

// B fragment (32x16 bf16, K x N), B staged transposed in LDS as [n][k]:
// lane L: n = L&15, half = L>>4; elems 0..7 -> K = 8*half + i ; elems 8..15 -> K = 8*half + 16 + i
__device__ __forceinline__ v16bf frag_b(const __bf16* s /* &Bs[nb*LDB] */, int lane) {
  const int n = lane & 15, half = lane >> 4;
  const __bf16* p = s + n * LDB + half * 8;
  v8bf lo = *(const v8bf*)(p);
  v8bf hi = *(const v8bf*)(p + 16);
  v16bf b;
#pragma unroll
  for (int i = 0; i < 8; ++i) { b[i] = lo[i]; b[i + 8] = hi[i]; }
  return b;
}

__device__ __forceinline__ int find_expert(const int* __restrict__ cnt, int row0) {
  int e = NE - 1, off = 0;
#pragma unroll
  for (int i = 0; i < NE; ++i) {
    int c = cnt[i];
    if (row0 < off + c) { e = i; break; }
    off += c;
  }
  return e;
}

// ---------------- staging helpers ----------------

// 128x32 fp32 -> bf16 row-major tile
__device__ __forceinline__ void stage_a_f32(const float* __restrict__ src, size_t ld,
                                            int k0, int tid, unsigned short* AsU) {
#pragma unroll
  for (int r = 0; r < 4; ++r) {
    const int row = (tid >> 3) + (r << 5);
    const int c4  = (tid & 7) << 2;
    float4 v = *(const float4*)(src + (size_t)row * ld + k0 + c4);
    ushort4 h; h.x = f2bfu(v.x); h.y = f2bfu(v.y); h.z = f2bfu(v.z); h.w = f2bfu(v.w);
    *(ushort4*)(AsU + row * LDA + c4) = h;
  }
}

// 32x64 fp32 -> bf16 transposed tile ([n][k] in LDS)
__device__ __forceinline__ void stage_b_f32t(const float* __restrict__ src, size_t ld,
                                             int k0, int nbase, int tid,
                                             unsigned short* BsU) {
#pragma unroll
  for (int r = 0; r < 2; ++r) {
    const int k  = (tid >> 4) + (r << 4);
    const int n4 = (tid & 15) << 2;
    float4 v = *(const float4*)(src + (size_t)(k0 + k) * ld + nbase + n4);
    BsU[(n4 + 0) * LDB + k] = f2bfu(v.x);
    BsU[(n4 + 1) * LDB + k] = f2bfu(v.y);
    BsU[(n4 + 2) * LDB + k] = f2bfu(v.z);
    BsU[(n4 + 3) * LDB + k] = f2bfu(v.w);
  }
}

// TDM: DMA a 128-row x 32-bf16 (64B) tile from global into LDS, padding each
// row from 32 to LDA(=40) halves via the TDM pad feature (pad_interval=16 DW
// of data, pad_amount=4 DW). One issue per block; tracked with TENSORcnt.
__device__ __forceinline__ void tdm_load_tile_a(const unsigned short* gsrc,
                                                unsigned lds_off) {
  const unsigned long long ga = (unsigned long long)(uintptr_t)gsrc;
  u32x4 g0;
  g0[0] = 1u;                               // count=1 valid descriptor (user mode)
  g0[1] = lds_off;                          // lds_addr (bytes)
  g0[2] = (unsigned)ga;                     // global_addr[31:0]
  g0[3] = (unsigned)(ga >> 32) | 0x80000000u; // global_addr[56:32] | type=2 (bits 127:126)
  i32x8 g1;
  g1[0] = (1 << 16)                         // data_size = 1 -> 2 bytes/element
        | (1 << 20)                         // pad_enable (LDS destination padding)
        | (3 << 22)                         // pad_interval code 3 = 16 DWORDs (64B) of data
        | (3 << 25);                        // pad_amount  code 3 = 4 DWORDs (16B = 8 halves)
  g1[1] = (KT << 16);                       // tensor_dim0 = 32 elements (tile-local view)
  g1[2] = (TM << 16);                       // tensor_dim1 = 128 rows
  g1[3] = (KT << 16);                       // tile_dim0 = 32 elements (64B row)
  g1[4] = TM;                               // tile_dim1 = 128 rows
  g1[5] = FFN;                              // tensor_dim0_stride = 1408 elements
  g1[6] = 0;
  g1[7] = 0;
  i32x4 z4; z4[0] = 0; z4[1] = 0; z4[2] = 0; z4[3] = 0;  // groups 2/3 unused (2D tile)
  i32x8 z8;
#pragma unroll
  for (int i = 0; i < 8; ++i) z8[i] = 0;
  __builtin_amdgcn_tensor_load_to_lds(g0, g1, z4, z4, z8, 0);
}

// ---------------- Kernel 1: act = swiglu(x @ w1[e]) -> bf16 ----------------
__global__ __launch_bounds__(256)
void moe_gemm1_swiglu(const float* __restrict__ X,
                      const int* __restrict__ cnt,
                      const float* __restrict__ W1,
                      unsigned short* __restrict__ ACT) {
  __shared__ __align__(16) __bf16 As[2][TM * LDA];
  __shared__ __align__(16) __bf16 Bg[2][TN * LDB];
  __shared__ __align__(16) __bf16 Bu[2][TN * LDB];

  const int tid  = threadIdx.x;
  const int lane = tid & 31;
  const int wave = tid >> 5;
  const int wm = (wave & 3) << 5;     // 0,32,64,96
  const int wn = (wave >> 2) << 5;    // 0,32
  const int jbase = blockIdx.x * TN;  // [0, FFN)
  const int row0  = blockIdx.y * TM;

  const int e = find_expert(cnt, row0);
  const float* W1e = W1 + (size_t)e * HIDDEN * FFN2;
  const float* Xr  = X + (size_t)row0 * HIDDEN;

  v8f accG[2][2], accU[2][2];
#pragma unroll
  for (int i = 0; i < 2; ++i)
#pragma unroll
    for (int j = 0; j < 2; ++j) { accG[i][j] = (v8f){}; accU[i][j] = (v8f){}; }

  const int nk = HIDDEN / KT;  // 64

  // prologue: stage tile 0 into buffer 0
  stage_a_f32(Xr, HIDDEN, 0, tid, (unsigned short*)As[0]);
  stage_b_f32t(W1e, FFN2, 0, jbase, tid, (unsigned short*)Bg[0]);
  stage_b_f32t(W1e + FFN, FFN2, 0, jbase, tid, (unsigned short*)Bu[0]);

  for (int kt = 0; kt < nk; ++kt) {
    __syncthreads();
    const int cur = kt & 1;
    if (kt + 1 < nk) {
      const int k0n = (kt + 1) * KT;
      stage_a_f32(Xr, HIDDEN, k0n, tid, (unsigned short*)As[cur ^ 1]);
      stage_b_f32t(W1e, FFN2, k0n, jbase, tid, (unsigned short*)Bg[cur ^ 1]);
      stage_b_f32t(W1e + FFN, FFN2, k0n, jbase, tid, (unsigned short*)Bu[cur ^ 1]);
    }

    v16bf a0 = frag_a(As[cur] + (wm +  0) * LDA, lane);
    v16bf a1 = frag_a(As[cur] + (wm + 16) * LDA, lane);
    v16bf g0 = frag_b(Bg[cur] + (wn +  0) * LDB, lane);
    v16bf g1 = frag_b(Bg[cur] + (wn + 16) * LDB, lane);
    v16bf u0 = frag_b(Bu[cur] + (wn +  0) * LDB, lane);
    v16bf u1 = frag_b(Bu[cur] + (wn + 16) * LDB, lane);

    accG[0][0] = __builtin_amdgcn_wmma_f32_16x16x32_bf16(false, a0, false, g0, (short)0, accG[0][0], false, false);
    accG[0][1] = __builtin_amdgcn_wmma_f32_16x16x32_bf16(false, a0, false, g1, (short)0, accG[0][1], false, false);
    accG[1][0] = __builtin_amdgcn_wmma_f32_16x16x32_bf16(false, a1, false, g0, (short)0, accG[1][0], false, false);
    accG[1][1] = __builtin_amdgcn_wmma_f32_16x16x32_bf16(false, a1, false, g1, (short)0, accG[1][1], false, false);
    accU[0][0] = __builtin_amdgcn_wmma_f32_16x16x32_bf16(false, a0, false, u0, (short)0, accU[0][0], false, false);
    accU[0][1] = __builtin_amdgcn_wmma_f32_16x16x32_bf16(false, a0, false, u1, (short)0, accU[0][1], false, false);
    accU[1][0] = __builtin_amdgcn_wmma_f32_16x16x32_bf16(false, a1, false, u0, (short)0, accU[1][0], false, false);
    accU[1][1] = __builtin_amdgcn_wmma_f32_16x16x32_bf16(false, a1, false, u1, (short)0, accU[1][1], false, false);
  }

  // Epilogue: act = silu(g) * u, stored bf16. D layout: elem r -> M = r + 8*(lane>>4), N = lane&15
  const int nlo = lane & 15, half = lane >> 4;
#pragma unroll
  for (int fm = 0; fm < 2; ++fm) {
#pragma unroll
    for (int fn = 0; fn < 2; ++fn) {
      v8f g = accG[fm][fn], u = accU[fm][fn];
      const int col = jbase + wn + (fn << 4) + nlo;
#pragma unroll
      for (int r = 0; r < 8; ++r) {
        const int row = row0 + wm + (fm << 4) + (half << 3) + r;
        float gv = g[r], uv = u[r];
        // fast silu: g * rcp(1+exp(-g)) -> v_exp_f32 + v_rcp_f32 + v_mul
        float s = gv * __builtin_amdgcn_rcpf(1.0f + __expf(-gv));
        ACT[(size_t)row * FFN + col] = f2bfu(s * uv);
      }
    }
  }
}

// ---------------- Kernel 2: out = act @ w2[e] -> fp32 ----------------
__global__ __launch_bounds__(256)
void moe_gemm2(const unsigned short* __restrict__ ACT,
               const int* __restrict__ cnt,
               const float* __restrict__ W2,
               float* __restrict__ OUT) {
  __shared__ __align__(16) __bf16 As[2][TM * LDA];
  __shared__ __align__(16) __bf16 Bs[2][TN * LDB];

  const int tid  = threadIdx.x;
  const int lane = tid & 31;
  const int wave = tid >> 5;
  const int wm = (wave & 3) << 5;
  const int wn = (wave >> 2) << 5;
  const int nbase = blockIdx.x * TN;  // [0, HIDDEN)
  const int row0  = blockIdx.y * TM;

  const int e = find_expert(cnt, row0);
  const float* W2e = W2 + (size_t)e * FFN * HIDDEN;
  const unsigned short* Ar = ACT + (size_t)row0 * FFN;

  const unsigned ldsA0 = lds_offset_of(&As[0][0]);
  const unsigned ldsA1 = lds_offset_of(&As[1][0]);

  v8f acc[2][2];
#pragma unroll
  for (int i = 0; i < 2; ++i)
#pragma unroll
    for (int j = 0; j < 2; ++j) acc[i][j] = (v8f){};

  const int nk = FFN / KT;  // 44

  // prologue: A tile via Tensor Data Mover (pure bf16 copy), B tile via VALU convert
  if (wave == 0) tdm_load_tile_a(Ar, ldsA0);
  stage_b_f32t(W2e, HIDDEN, 0, nbase, tid, (unsigned short*)Bs[0]);

  for (int kt = 0; kt < nk; ++kt) {
    if (wave == 0) __builtin_amdgcn_s_wait_tensorcnt(0);  // A[cur] DMA landed
    __syncthreads();
    const int cur = kt & 1;
    if (kt + 1 < nk) {
      if (wave == 0)
        tdm_load_tile_a(Ar + (kt + 1) * KT, cur ? ldsA0 : ldsA1);
      stage_b_f32t(W2e, HIDDEN, (kt + 1) * KT, nbase, tid, (unsigned short*)Bs[cur ^ 1]);
    }

    v16bf a0 = frag_a(As[cur] + (wm +  0) * LDA, lane);
    v16bf a1 = frag_a(As[cur] + (wm + 16) * LDA, lane);
    v16bf b0 = frag_b(Bs[cur] + (wn +  0) * LDB, lane);
    v16bf b1 = frag_b(Bs[cur] + (wn + 16) * LDB, lane);

    acc[0][0] = __builtin_amdgcn_wmma_f32_16x16x32_bf16(false, a0, false, b0, (short)0, acc[0][0], false, false);
    acc[0][1] = __builtin_amdgcn_wmma_f32_16x16x32_bf16(false, a0, false, b1, (short)0, acc[0][1], false, false);
    acc[1][0] = __builtin_amdgcn_wmma_f32_16x16x32_bf16(false, a1, false, b0, (short)0, acc[1][0], false, false);
    acc[1][1] = __builtin_amdgcn_wmma_f32_16x16x32_bf16(false, a1, false, b1, (short)0, acc[1][1], false, false);
  }

  const int nlo = lane & 15, half = lane >> 4;
#pragma unroll
  for (int fm = 0; fm < 2; ++fm) {
#pragma unroll
    for (int fn = 0; fn < 2; ++fn) {
      v8f c = acc[fm][fn];
      const int col = nbase + wn + (fn << 4) + nlo;
#pragma unroll
      for (int r = 0; r < 8; ++r) {
        const int row = row0 + wm + (fm << 4) + (half << 3) + r;
        OUT[(size_t)row * HIDDEN + col] = c[r];
      }
    }
  }
}

extern "C" void kernel_launch(void* const* d_in, const int* in_sizes, int n_in,
                              void* d_out, int out_size, void* d_ws, size_t ws_size,
                              hipStream_t stream) {
  const float* X   = (const float*)d_in[0];           // [TOKENS, HIDDEN] fp32
  const int*   cnt = (const int*)d_in[1];             // [NE] tokens per expert
  const float* W1  = (const float*)d_in[2];           // [NE, HIDDEN, 2*FFN] fp32
  const float* W2  = (const float*)d_in[3];           // [NE, FFN, HIDDEN] fp32
  float* out = (float*)d_out;                         // [TOKENS, HIDDEN] fp32
  unsigned short* act = (unsigned short*)d_ws;        // [TOKENS, FFN] bf16 (46 MB)

  const int tokens = in_sizes[0] / HIDDEN;            // 16384
  dim3 blk(256);
  dim3 g1(FFN / TN, tokens / TM);                     // 22 x 128
  dim3 g2(HIDDEN / TN, tokens / TM);                  // 32 x 128

  hipLaunchKernelGGL(moe_gemm1_swiglu, g1, blk, 0, stream, X, cnt, W1, act);
  hipLaunchKernelGGL(moe_gemm2, g2, blk, 0, stream, act, cnt, W2, out);
}
